// TemporalAttention_1125281432070
// MI455X (gfx1250) — matrix operand order repaired and checked by reading, not compiled, not run on hardware
//
#include <hip/hip_runtime.h>
#include <hip/hip_bf16.h>

typedef __bf16 bf16;
typedef __attribute__((ext_vector_type(16))) __bf16 v16bf;
typedef __attribute__((ext_vector_type(8)))  __bf16 v8bf;
typedef __attribute__((ext_vector_type(8)))  float  v8f;

#define DIMC   1024
#define NHC    16
#define HDC    64
#define FC     16
#define SC     1472
#define MTOK   (FC * SC)      // 23552 token rows
#define WIN    8

// ---------------------------------------------------------------------------
// Stage 0: f32 -> bf16 conversion (x and weights). n must be a multiple of 4.
// ---------------------------------------------------------------------------
__global__ __launch_bounds__(256) void cvt_f32_bf16(const float* __restrict__ in,
                                                    bf16* __restrict__ out, int n) {
  int i = (blockIdx.x * 256 + threadIdx.x) * 4;
  if (i < n) {
    float4 v = *(const float4*)(in + i);
    out[i + 0] = (bf16)v.x;
    out[i + 1] = (bf16)v.y;
    out[i + 2] = (bf16)v.z;
    out[i + 3] = (bf16)v.w;
  }
}

// ---------------------------------------------------------------------------
// WMMA GEMM:  C[M,N] = A[M,K] @ W[N,K]^T + bias   (bf16 in, f32 accumulate)
// Block = 256 threads = 8 waves. Wave tile: 32(M) x 64(N) -> 8 WMMA / K-step
// with each B fragment reused by 2 WMMAs (1.5 b128 loads per WMMA).
// Block tile: 32 x 512. Grid: (N/512, M/32).
// Fragment layouts used here (contiguous K order, same in A and B):
//   A (16x32 bf16): lane m = L&15, half = L>>4; elem i -> K = half*16 + i
//   B (32x16 bf16): lane n = L&15, half = L>>4; elem i -> K = half*16 + i
//   C (16x16 f32):  lane n = L&15; vgpr r -> row m = r + 8*half
// ---------------------------------------------------------------------------
template <typename OutT>
__global__ __launch_bounds__(256) void gemm_wmma_bf16(
    const bf16* __restrict__ A, const bf16* __restrict__ W,
    const float* __restrict__ bias, OutT* __restrict__ C,
    int M, int N, int K) {
  const int lane = threadIdx.x & 31;
  const int wave = threadIdx.x >> 5;
  const int m0   = blockIdx.y * 32;
  const int n0   = blockIdx.x * 512 + wave * 64;
  const int lr   = lane & 15;   // row for A-frag / col for B-frag & C-frag
  const int half = lane >> 4;

  const v8f z8 = {0.f, 0.f, 0.f, 0.f, 0.f, 0.f, 0.f, 0.f};
  v8f acc00 = z8, acc01 = z8, acc02 = z8, acc03 = z8;  // rows m0..m0+15
  v8f acc10 = z8, acc11 = z8, acc12 = z8, acc13 = z8;  // rows m0+16..m0+31

  const bf16* __restrict__ ar0 = A + (size_t)(m0 + lr) * K + half * 16;
  const bf16* __restrict__ ar1 = A + (size_t)(m0 + 16 + lr) * K + half * 16;
  const bf16* __restrict__ b0  = W + (size_t)(n0 + 0  + lr) * K + half * 16;
  const bf16* __restrict__ b1  = W + (size_t)(n0 + 16 + lr) * K + half * 16;
  const bf16* __restrict__ b2  = W + (size_t)(n0 + 32 + lr) * K + half * 16;
  const bf16* __restrict__ b3  = W + (size_t)(n0 + 48 + lr) * K + half * 16;

  for (int k0 = 0; k0 < K; k0 += 32) {
    // hint-prefetch ahead (speculative; emits global_prefetch_b8)
    __builtin_prefetch(ar0 + k0 + 128, 0, 1);
    __builtin_prefetch(b0 + k0 + 128, 0, 1);

    v8bf a0l = *(const v8bf*)(ar0 + k0);
    v8bf a0h = *(const v8bf*)(ar0 + k0 + 8);
    v8bf a1l = *(const v8bf*)(ar1 + k0);
    v8bf a1h = *(const v8bf*)(ar1 + k0 + 8);
    v16bf a0 = __builtin_shufflevector(a0l, a0h, 0,1,2,3,4,5,6,7,8,9,10,11,12,13,14,15);
    v16bf a1 = __builtin_shufflevector(a1l, a1h, 0,1,2,3,4,5,6,7,8,9,10,11,12,13,14,15);

    v8bf l0 = *(const v8bf*)(b0 + k0); v8bf h0 = *(const v8bf*)(b0 + k0 + 8);
    v8bf l1 = *(const v8bf*)(b1 + k0); v8bf h1 = *(const v8bf*)(b1 + k0 + 8);
    v8bf l2 = *(const v8bf*)(b2 + k0); v8bf h2 = *(const v8bf*)(b2 + k0 + 8);
    v8bf l3 = *(const v8bf*)(b3 + k0); v8bf h3 = *(const v8bf*)(b3 + k0 + 8);
    v16bf f0 = __builtin_shufflevector(l0, h0, 0,1,2,3,4,5,6,7,8,9,10,11,12,13,14,15);
    v16bf f1 = __builtin_shufflevector(l1, h1, 0,1,2,3,4,5,6,7,8,9,10,11,12,13,14,15);
    v16bf f2 = __builtin_shufflevector(l2, h2, 0,1,2,3,4,5,6,7,8,9,10,11,12,13,14,15);
    v16bf f3 = __builtin_shufflevector(l3, h3, 0,1,2,3,4,5,6,7,8,9,10,11,12,13,14,15);

    acc00 = __builtin_amdgcn_wmma_f32_16x16x32_bf16(false, a0, false, f0, (short)0, acc00, false, false);
    acc10 = __builtin_amdgcn_wmma_f32_16x16x32_bf16(false, a1, false, f0, (short)0, acc10, false, false);
    acc01 = __builtin_amdgcn_wmma_f32_16x16x32_bf16(false, a0, false, f1, (short)0, acc01, false, false);
    acc11 = __builtin_amdgcn_wmma_f32_16x16x32_bf16(false, a1, false, f1, (short)0, acc11, false, false);
    acc02 = __builtin_amdgcn_wmma_f32_16x16x32_bf16(false, a0, false, f2, (short)0, acc02, false, false);
    acc12 = __builtin_amdgcn_wmma_f32_16x16x32_bf16(false, a1, false, f2, (short)0, acc12, false, false);
    acc03 = __builtin_amdgcn_wmma_f32_16x16x32_bf16(false, a0, false, f3, (short)0, acc03, false, false);
    acc13 = __builtin_amdgcn_wmma_f32_16x16x32_bf16(false, a1, false, f3, (short)0, acc13, false, false);
  }

  v8f accs[2][4] = {{acc00, acc01, acc02, acc03}, {acc10, acc11, acc12, acc13}};
#pragma unroll
  for (int mi = 0; mi < 2; ++mi) {
#pragma unroll
    for (int t = 0; t < 4; ++t) {
      int n = n0 + t * 16 + lr;
      float bv = bias[n];
#pragma unroll
      for (int r = 0; r < 8; ++r) {
        int m = m0 + mi * 16 + r + 8 * half;
        C[(size_t)m * N + n] = (OutT)(accs[mi][t][r] + bv);
      }
    }
  }
}

// ---------------------------------------------------------------------------
// Stage 2: fused RMSNorm + RoPE, in place on Q / K (bf16).
// Row index == RoPE position (the (f,s) flatten is the row order).
// Grid: (MTOK, 2) ; 256 threads ; 4 elements (= 2 RoPE pairs) per thread.
// ---------------------------------------------------------------------------
__global__ __launch_bounds__(256) void norm_rope(
    bf16* __restrict__ Qb, bf16* __restrict__ Kb,
    const float* __restrict__ fcos, const float* __restrict__ fsin,
    const float* __restrict__ nqw, const float* __restrict__ nkw) {
  const int row = blockIdx.x;
  bf16* __restrict__ rp = ((blockIdx.y == 0) ? Qb : Kb) + (size_t)row * DIMC;
  const float* __restrict__ w = (blockIdx.y == 0) ? nqw : nkw;

  const int base = threadIdx.x * 4;
  float v0 = (float)rp[base + 0], v1 = (float)rp[base + 1];
  float v2 = (float)rp[base + 2], v3 = (float)rp[base + 3];

  __shared__ float red[256];
  red[threadIdx.x] = v0 * v0 + v1 * v1 + v2 * v2 + v3 * v3;
  __syncthreads();
  for (int st = 128; st > 0; st >>= 1) {
    if (threadIdx.x < st) red[threadIdx.x] += red[threadIdx.x + st];
    __syncthreads();
  }
  const float rms = rsqrtf(red[0] * (1.0f / DIMC) + 1e-6f);

  float vv[4] = {v0, v1, v2, v3};
#pragma unroll
  for (int j = 0; j < 2; ++j) {
    int pi = (base >> 1) + j;        // global even/odd pair index, 0..511
    int fi = pi & 31;                // freq index within head (HD/2 = 32)
    float c = fcos[(size_t)row * 32 + fi];
    float s = fsin[(size_t)row * 32 + fi];
    float xr = vv[2 * j + 0] * rms * w[base + 2 * j + 0];
    float xi = vv[2 * j + 1] * rms * w[base + 2 * j + 1];
    rp[base + 2 * j + 0] = (bf16)(xr * c - xi * s);
    rp[base + 2 * j + 1] = (bf16)(xr * s + xi * c);
  }
}

// ---------------------------------------------------------------------------
// Stage 3: temporal attention over f=16 frames, causal window 8.
// One block per (head, s). LDS-staged 16x64 Q/K/V tiles; softmax in LDS.
// Output written in (f, s, DIM) order -> row f*SC+s (direct GEMM-4 input).
// ---------------------------------------------------------------------------
__global__ __launch_bounds__(256) void attn_window(
    const bf16* __restrict__ Qb, const bf16* __restrict__ Kb,
    const bf16* __restrict__ Vb, bf16* __restrict__ AO) {
  const int h = blockIdx.x;
  const int s = blockIdx.y;
  __shared__ float Qs[FC][HDC];
  __shared__ float Ks[FC][HDC];
  __shared__ float Vs[FC][HDC];
  __shared__ float P[FC][FC + 1];

  for (int idx = threadIdx.x; idx < FC * HDC; idx += 256) {
    int f = idx >> 6, d = idx & 63;
    size_t off = (size_t)(f * SC + s) * DIMC + h * HDC + d;
    Qs[f][d] = (float)Qb[off];
    Ks[f][d] = (float)Kb[off];
    Vs[f][d] = (float)Vb[off];
  }
  __syncthreads();

  {  // scores (scaled, masked)
    int q = threadIdx.x >> 4, k = threadIdx.x & 15;
    float sc = -1e30f;
    if (k <= q && (q - k) < WIN) {
      float acc = 0.f;
#pragma unroll
      for (int d = 0; d < HDC; ++d) acc += Qs[q][d] * Ks[k][d];
      sc = acc * 0.125f;  // 1/sqrt(64)
    }
    P[q][k] = sc;
  }
  __syncthreads();

  {  // softmax per row (each thread owns one (q,k) entry)
    int q = threadIdx.x >> 4, k = threadIdx.x & 15;
    float mx = -1e30f;
#pragma unroll
    for (int j = 0; j < FC; ++j) mx = fmaxf(mx, P[q][j]);
    float sum = 0.f;
#pragma unroll
    for (int j = 0; j < FC; ++j) sum += __expf(P[q][j] - mx);
    float pv = __expf(P[q][k] - mx) / sum;
    __syncthreads();  // all reads done before overwrite
    P[q][k] = pv;
  }
  __syncthreads();

  for (int idx = threadIdx.x; idx < FC * HDC; idx += 256) {
    int f = idx >> 6, d = idx & 63;
    float acc = 0.f;
#pragma unroll
    for (int k = 0; k < FC; ++k) acc += P[f][k] * Vs[k][d];
    AO[(size_t)(f * SC + s) * DIMC + h * HDC + d] = (bf16)acc;
  }
}

// ---------------------------------------------------------------------------
extern "C" void kernel_launch(void* const* d_in, const int* in_sizes, int n_in,
                              void* d_out, int out_size, void* d_ws, size_t ws_size,
                              hipStream_t stream) {
  const float* x    = (const float*)d_in[0];
  const float* fcos = (const float*)d_in[1];
  const float* fsin = (const float*)d_in[2];
  const float* q_w  = (const float*)d_in[3];
  const float* q_b  = (const float*)d_in[4];
  const float* k_w  = (const float*)d_in[5];
  const float* k_b  = (const float*)d_in[6];
  const float* v_w  = (const float*)d_in[7];
  const float* v_b  = (const float*)d_in[8];
  const float* o_w  = (const float*)d_in[9];
  const float* o_b  = (const float*)d_in[10];
  const float* nqw  = (const float*)d_in[11];
  const float* nkw  = (const float*)d_in[12];
  float* out = (float*)d_out;

  char*  ws  = (char*)d_ws;
  size_t off = 0;
  auto alloc = [&](size_t bytes) -> void* {
    void* p = ws + off;
    off = (off + bytes + 255) & ~(size_t)255;
    return p;
  };
  const size_t tokb = (size_t)MTOK * DIMC * sizeof(bf16);
  const size_t wb   = (size_t)DIMC * DIMC * sizeof(bf16);
  bf16* xb  = (bf16*)alloc(tokb);
  bf16* wqb = (bf16*)alloc(wb);
  bf16* wkb = (bf16*)alloc(wb);
  bf16* wvb = (bf16*)alloc(wb);
  bf16* wob = (bf16*)alloc(wb);
  bf16* Qb  = (bf16*)alloc(tokb);
  bf16* Kb  = (bf16*)alloc(tokb);
  bf16* Vb  = (bf16*)alloc(tokb);
  bf16* AOb = (bf16*)alloc(tokb);

  // Stage 0: conversions
  const int nx = MTOK * DIMC, nw = DIMC * DIMC;
  cvt_f32_bf16<<<dim3((nx / 4 + 255) / 256), dim3(256), 0, stream>>>(x, xb, nx);
  cvt_f32_bf16<<<dim3((nw / 4 + 255) / 256), dim3(256), 0, stream>>>(q_w, wqb, nw);
  cvt_f32_bf16<<<dim3((nw / 4 + 255) / 256), dim3(256), 0, stream>>>(k_w, wkb, nw);
  cvt_f32_bf16<<<dim3((nw / 4 + 255) / 256), dim3(256), 0, stream>>>(v_w, wvb, nw);
  cvt_f32_bf16<<<dim3((nw / 4 + 255) / 256), dim3(256), 0, stream>>>(o_w, wob, nw);

  // Stage 1: QKV projections (WMMA bf16)
  dim3 gg(DIMC / 512, MTOK / 32), gb(256);
  gemm_wmma_bf16<bf16><<<gg, gb, 0, stream>>>(xb, wqb, q_b, Qb, MTOK, DIMC, DIMC);
  gemm_wmma_bf16<bf16><<<gg, gb, 0, stream>>>(xb, wkb, k_b, Kb, MTOK, DIMC, DIMC);
  gemm_wmma_bf16<bf16><<<gg, gb, 0, stream>>>(xb, wvb, v_b, Vb, MTOK, DIMC, DIMC);

  // Stage 2: RMSNorm + RoPE on Q and K
  norm_rope<<<dim3(MTOK, 2), dim3(256), 0, stream>>>(Qb, Kb, fcos, fsin, nqw, nkw);

  // Stage 3: windowed temporal attention
  attn_window<<<dim3(NHC, SC), dim3(256), 0, stream>>>(Qb, Kb, Vb, AOb);

  // Stage 4: output projection (WMMA bf16 -> f32 out)
  gemm_wmma_bf16<float><<<gg, gb, 0, stream>>>(AOb, wob, o_b, out, MTOK, DIMC, DIMC);
}
